// NeuralODEForecast_47467978555437
// MI455X (gfx1250) — matrix operand order repaired and checked by reading, not compiled, not run on hardware
//
#include <hip/hip_runtime.h>

// ---------- types ----------
typedef __attribute__((ext_vector_type(16))) __bf16 v16bf;
typedef __attribute__((ext_vector_type(8)))  __bf16 v8bf;
typedef __attribute__((ext_vector_type(8)))  float  v8f;

// ---------- problem constants ----------
constexpr int T  = 256;
constexpr int BB = 4096;
constexpr int D  = 32;
constexpr int H  = 256;
constexpr int L  = 32;
constexpr int HO = 512;
constexpr int OUTD = 8;
constexpr float DELTA = 1.0f;

constexpr int KC = 320;            // packed K: 64 (x | dt | pad) + 256 (h)
constexpr int GRU_STRIDE = KC + 8; // LDS row stride in halves (bank-conflict pad)

// ---------- helpers ----------
__device__ __forceinline__ __bf16 f2bf(float x) {
  union { float f; unsigned u; } v; v.f = x;
  unsigned r = (v.u + 0x7FFFu + ((v.u >> 16) & 1u)) >> 16;
  union { unsigned short s; __bf16 b; } o; o.s = (unsigned short)r;
  return o.b;
}
__device__ __forceinline__ float bf2f(__bf16 x) {
  union { unsigned short s; __bf16 b; } i; i.b = x;
  union { float f; unsigned u; } v; v.u = ((unsigned)i.s) << 16;
  return v.f;
}
__device__ __forceinline__ float sigmoid_(float x) { return 1.0f / (1.0f + __expf(-x)); }
__device__ __forceinline__ float tanh_(float x) { float e = __expf(2.0f * x); return (e - 1.0f) / (e + 1.0f); }

__device__ __forceinline__ v8f wmma_bf16(v16bf a, v16bf b, v8f c) {
  return __builtin_amdgcn_wmma_f32_16x16x32_bf16(false, a, false, b, (short)0, c, false, false);
}

union AFrag { v16bf v; v8bf h[2]; };

// =====================================================================
// Prep: transpose/pack all weights to N-major bf16; zero h state.
// Wcat_t[n][k], n in [0,768): k<33 -> W_ih[k][n]; 33..63 -> 0; >=64 -> W_hh[k-64][n]
// =====================================================================
__global__ void prep_kernel(const float* __restrict__ W_ih, const float* __restrict__ W_hh,
                            const float* __restrict__ W_lat, const float* __restrict__ W1,
                            const float* __restrict__ W2, const float* __restrict__ W3,
                            const float* __restrict__ Wd1,
                            __bf16* Wcat_t, __bf16* Wlat_t, __bf16* W1t,
                            __bf16* W2t, __bf16* W3t, __bf16* Wd1t,
                            __bf16* h0b, float* h0f) {
  int tid = blockIdx.x * blockDim.x + threadIdx.x;
  int np  = gridDim.x * blockDim.x;
  for (int i = tid; i < 768 * KC; i += np) {
    int n = i / KC, k = i % KC;
    float v;
    if (k < 33)      v = W_ih[k * 768 + n];
    else if (k < 64) v = 0.0f;
    else             v = W_hh[(k - 64) * 768 + n];
    Wcat_t[i] = f2bf(v);
  }
  for (int i = tid; i < 32 * 256;  i += np) { int n = i / 256, k = i % 256; Wlat_t[i] = f2bf(W_lat[k * 64  + n]); }
  for (int i = tid; i < 512 * 32;  i += np) { int n = i / 32,  k = i % 32;  W1t[i]   = f2bf(W1[k * 512 + n]); }
  for (int i = tid; i < 512 * 512; i += np) { int n = i / 512, k = i % 512; W2t[i]   = f2bf(W2[k * 512 + n]); }
  for (int i = tid; i < 32 * 512;  i += np) { int n = i / 512, k = i % 512; W3t[i]   = f2bf(W3[k * 32  + n]); }
  for (int i = tid; i < 256 * 32;  i += np) { int n = i / 32,  k = i % 32;  Wd1t[i]  = f2bf(Wd1[k * 256 + n]); }
  for (int i = tid; i < BB * H;    i += np) { h0b[i] = f2bf(0.0f); h0f[i] = 0.0f; }
}

// =====================================================================
// One GRU step. Grid (BB/128, H/32), 128 threads = 4 waves.
// Each wave: 32 rows x 32 h-cols (2 M-subtiles x 2 N-subtiles, 3 gates,
// n-gate split into x-part/h-part) -> 16 v8f accumulators.
// LDS: A = [x | dt | pad | h_prev] (128 x GRU_STRIDE halves)
//      B = weight tile for this nb: 96 rows (3 gates x 32 cols) x GRU_STRIDE
// Register blocking: A frags reused across 2 ns, B frags across 2 ms
// -> 16 ds_load_b128 per 12 WMMAs.
// =====================================================================
__global__ __launch_bounds__(128) void gru_step_kernel(
    const float* __restrict__ x_hist, const float* __restrict__ t_hist,
    const __bf16* __restrict__ Wcat_t,
    const float* __restrict__ b_ih, const float* __restrict__ b_hh,
    const __bf16* __restrict__ h_in, __bf16* __restrict__ h_out,
    float* __restrict__ hf32, int t_idx) {
  extern __shared__ __bf16 lds[];
  __bf16* At = lds;                          // [128][GRU_STRIDE]
  __bf16* Bs = lds + 128 * GRU_STRIDE;       // [96][GRU_STRIDE]
  const int tid   = threadIdx.x;
  const int mb    = blockIdx.x;
  const int nb    = blockIdx.y;
  const int mbase = mb * 128;

  // stage h part of A (k = 64..319), coalesced 16B chunks
  for (int idx = tid; idx < 128 * 32; idx += 128) {
    int row = idx >> 5, kc = idx & 31;
    v8bf v = *reinterpret_cast<const v8bf*>(h_in + (size_t)(mbase + row) * H + kc * 8);
    *reinterpret_cast<v8bf*>(&At[row * GRU_STRIDE + 64 + kc * 8]) = v;
  }
  // stage B tile: rows 0..95 = gate g = row>>5, col = nb*32 + (row&31)
  for (int idx = tid; idx < 96 * 40; idx += 128) {
    int row = idx / 40, kc = idx % 40;
    int g = row >> 5, col = row & 31;
    v8bf v = *reinterpret_cast<const v8bf*>(Wcat_t + (size_t)(g * 256 + nb * 32 + col) * KC + kc * 8);
    *reinterpret_cast<v8bf*>(&Bs[row * GRU_STRIDE + kc * 8]) = v;
  }
  // zero pad region of A, k = 32..63 (k=32 overwritten with dt below)
  for (int idx = tid; idx < 128 * 4; idx += 128) {
    int row = idx >> 2, kc = idx & 3;
    unsigned long long* q = reinterpret_cast<unsigned long long*>(&At[row * GRU_STRIDE + 32 + kc * 8]);
    q[0] = 0ull; q[1] = 0ull;
  }
  // x part of A, k = 0..31
  for (int idx = tid; idx < 128 * 8; idx += 128) {
    int row = idx >> 3, kc = (idx & 7) * 4;
    const float* xp = x_hist + ((size_t)t_idx * BB + mbase + row) * D + kc;
    __bf16* dst = &At[row * GRU_STRIDE + kc];
    dst[0] = f2bf(xp[0]); dst[1] = f2bf(xp[1]); dst[2] = f2bf(xp[2]); dst[3] = f2bf(xp[3]);
  }
  // dt at k = 32 (sequence reversed; dt defined on original time index)
  {
    int row = tid;
    float dt = 0.0f;
    if (t_idx > 0)
      dt = t_hist[(size_t)t_idx * BB + mbase + row] - t_hist[(size_t)(t_idx - 1) * BB + mbase + row];
    At[row * GRU_STRIDE + 32] = f2bf(dt);
  }
  __syncthreads();

  const int lane  = tid & 31;
  const int wv    = tid >> 5;   // 0..3
  const int rbase = wv * 32;    // 32 rows per wave
  const int lhalf = lane >> 4;  // K-phase / M+8 select
  const int lmod  = lane & 15;  // row (A) / col (B,C)

  v8f acc_r[2][2]  = {};  // [ms][ns]
  v8f acc_z[2][2]  = {};
  v8f acc_in[2][2] = {};  // x-part of n gate (K 0..63)
  v8f acc_hn[2][2] = {};  // h-part of n gate (K 64..319)

#pragma unroll
  for (int kb = 0; kb < KC / 32; ++kb) {
    const int k0 = kb * 32;
    AFrag a[2];
#pragma unroll
    for (int ms = 0; ms < 2; ++ms) {
      const __bf16* ap = &At[(rbase + ms * 16 + lmod) * GRU_STRIDE + k0 + lhalf * 16];
      a[ms].h[0] = *reinterpret_cast<const v8bf*>(ap);
      a[ms].h[1] = *reinterpret_cast<const v8bf*>(ap + 8);
    }
    const int kf = k0 + lhalf * 16;
#pragma unroll
    for (int ns = 0; ns < 2; ++ns) {
      const int cl = ns * 16 + lmod;   // column within the 32-wide tile
      AFrag br, bz, bn;
      const __bf16* p0 = &Bs[(0 * 32 + cl) * GRU_STRIDE + kf];
      const __bf16* p1 = &Bs[(1 * 32 + cl) * GRU_STRIDE + kf];
      const __bf16* p2 = &Bs[(2 * 32 + cl) * GRU_STRIDE + kf];
      br.h[0] = *reinterpret_cast<const v8bf*>(p0); br.h[1] = *reinterpret_cast<const v8bf*>(p0 + 8);
      bz.h[0] = *reinterpret_cast<const v8bf*>(p1); bz.h[1] = *reinterpret_cast<const v8bf*>(p1 + 8);
      bn.h[0] = *reinterpret_cast<const v8bf*>(p2); bn.h[1] = *reinterpret_cast<const v8bf*>(p2 + 8);
#pragma unroll
      for (int ms = 0; ms < 2; ++ms) {
        acc_r[ms][ns] = wmma_bf16(a[ms].v, br.v, acc_r[ms][ns]);
        acc_z[ms][ns] = wmma_bf16(a[ms].v, bz.v, acc_z[ms][ns]);
        if (kb < 2) acc_in[ms][ns] = wmma_bf16(a[ms].v, bn.v, acc_in[ms][ns]);
        else        acc_hn[ms][ns] = wmma_bf16(a[ms].v, bn.v, acc_hn[ms][ns]);
      }
    }
  }

#pragma unroll
  for (int ns = 0; ns < 2; ++ns) {
    const int ncol = nb * 32 + ns * 16 + lmod;
    const float brc = b_ih[ncol]       + b_hh[ncol];
    const float bzc = b_ih[256 + ncol] + b_hh[256 + ncol];
    const float bin = b_ih[512 + ncol];
    const float bhn = b_hh[512 + ncol];
#pragma unroll
    for (int ms = 0; ms < 2; ++ms) {
#pragma unroll
      for (int r = 0; r < 8; ++r) {
        const int gm = mbase + rbase + ms * 16 + r + 8 * lhalf;
        const size_t off = (size_t)gm * H + ncol;
        const float hold = hf32[off];
        const float rg = sigmoid_(acc_r[ms][ns][r] + brc);
        const float zg = sigmoid_(acc_z[ms][ns][r] + bzc);
        const float ng = tanh_(acc_in[ms][ns][r] + bin + rg * (acc_hn[ms][ns][r] + bhn));
        const float hn = (1.0f - zg) * ng + zg * hold;
        hf32[off]  = hn;
        h_out[off] = f2bf(hn);
      }
    }
  }
}

// =====================================================================
// Generic bf16 WMMA GEMM: C = act(A @ Bt^T + bias)
// A: MxK bf16 row-major, Bt: NxK bf16 (pre-transposed). Grid (M/128, N/32).
// Both A tile and the 32-column B tile are staged in LDS.
// act: 0=none 1=tanh 2=relu. Writes f32 and/or bf16.
// =====================================================================
__global__ __launch_bounds__(256) void gemm_bf16_kernel(
    const __bf16* __restrict__ A, const __bf16* __restrict__ Bt,
    const float* __restrict__ bias, float* __restrict__ Cf, __bf16* __restrict__ Cb,
    int K, int N, int act) {
  extern __shared__ __bf16 lds[];
  const int stride = K + 8;
  __bf16* As = lds;                 // [128][stride]
  __bf16* Bs = lds + 128 * stride;  // [32][stride]
  const int tid   = threadIdx.x;
  const int mbase = blockIdx.x * 128;

  const int kch = K / 8;
  for (int idx = tid; idx < 128 * kch; idx += 256) {
    int row = idx / kch, kc = idx % kch;
    v8bf v = *reinterpret_cast<const v8bf*>(A + (size_t)(mbase + row) * K + kc * 8);
    *reinterpret_cast<v8bf*>(&As[row * stride + kc * 8]) = v;
  }
  for (int idx = tid; idx < 32 * kch; idx += 256) {
    int row = idx / kch, kc = idx % kch;
    v8bf v = *reinterpret_cast<const v8bf*>(Bt + (size_t)(blockIdx.y * 32 + row) * K + kc * 8);
    *reinterpret_cast<v8bf*>(&Bs[row * stride + kc * 8]) = v;
  }
  __syncthreads();

  const int lane = tid & 31, wv = tid >> 5;
  const int rbase = wv * 16, lhalf = lane >> 4, lmod = lane & 15;
  v8f acc[2] = {};

  for (int kb = 0; kb < K / 32; ++kb) {
    const int k0 = kb * 32;
    AFrag a;
    const __bf16* ap = &As[(rbase + lmod) * stride + k0 + lhalf * 16];
    a.h[0] = *reinterpret_cast<const v8bf*>(ap);
    a.h[1] = *reinterpret_cast<const v8bf*>(ap + 8);
    const int kf = k0 + lhalf * 16;
#pragma unroll
    for (int ns = 0; ns < 2; ++ns) {
      AFrag b;
      const __bf16* bp = &Bs[(ns * 16 + lmod) * stride + kf];
      b.h[0] = *reinterpret_cast<const v8bf*>(bp);
      b.h[1] = *reinterpret_cast<const v8bf*>(bp + 8);
      acc[ns] = wmma_bf16(a.v, b.v, acc[ns]);
    }
  }

#pragma unroll
  for (int ns = 0; ns < 2; ++ns) {
    const int n = blockIdx.y * 32 + ns * 16 + lmod;
    const float bv = bias ? bias[n] : 0.0f;
#pragma unroll
    for (int r = 0; r < 8; ++r) {
      const int gm = mbase + rbase + r + 8 * lhalf;
      float v = acc[ns][r] + bv;
      if (act == 1)      v = tanh_(v);
      else if (act == 2) v = fmaxf(v, 0.0f);
      const size_t off = (size_t)gm * N + n;
      if (Cf) Cf[off] = v;
      if (Cb) Cb[off] = f2bf(v);
    }
  }
}

// =====================================================================
// RK4 combine: z = z0 + a1*k1 + a2*k2 + a3*k3 + a4*k4  (f32 + bf16 out)
// =====================================================================
__global__ void combine_kernel(const float* __restrict__ z0,
                               const float* __restrict__ k1, const float* __restrict__ k2,
                               const float* __restrict__ k3, const float* __restrict__ k4,
                               float a1, float a2, float a3, float a4,
                               float* __restrict__ zf, __bf16* __restrict__ zb, int n) {
  int i = blockIdx.x * blockDim.x + threadIdx.x;
  if (i < n) {
    float v = z0[i] + a1 * k1[i] + a2 * k2[i] + a3 * k3[i] + a4 * k4[i];
    zf[i] = v;
    zb[i] = f2bf(v);
  }
}

// =====================================================================
// Final decoder layer: out = d1 @ Wd2 + bd2  (N=8 too small for WMMA)
// =====================================================================
__global__ void decode2_kernel(const __bf16* __restrict__ d1, const float* __restrict__ Wd2,
                               const float* __restrict__ bd2, float* __restrict__ out) {
  int i = blockIdx.x * blockDim.x + threadIdx.x;
  if (i < BB * OUTD) {
    int b = i >> 3, o = i & 7;
    float acc = bd2[o];
    const __bf16* dp = d1 + (size_t)b * H;
    for (int k = 0; k < H; ++k) acc += bf2f(dp[k]) * Wd2[k * OUTD + o];
    out[i] = acc;
  }
}

// =====================================================================
extern "C" void kernel_launch(void* const* d_in, const int* in_sizes, int n_in,
                              void* d_out, int out_size, void* d_ws, size_t ws_size,
                              hipStream_t stream) {
  (void)in_sizes; (void)n_in; (void)out_size; (void)ws_size;
  const float* x_hist = (const float*)d_in[0];
  const float* t_hist = (const float*)d_in[1];
  const float* W_ih   = (const float*)d_in[2];
  const float* W_hh   = (const float*)d_in[3];
  const float* b_ih   = (const float*)d_in[4];
  const float* b_hh   = (const float*)d_in[5];
  const float* W_lat  = (const float*)d_in[6];
  const float* b_lat  = (const float*)d_in[7];
  const float* W1     = (const float*)d_in[8];
  const float* b1     = (const float*)d_in[9];
  const float* W2     = (const float*)d_in[10];
  const float* b2     = (const float*)d_in[11];
  const float* W3     = (const float*)d_in[12];
  const float* b3     = (const float*)d_in[13];
  const float* Wd1    = (const float*)d_in[14];
  const float* bd1    = (const float*)d_in[15];
  const float* Wd2    = (const float*)d_in[16];
  const float* bd2    = (const float*)d_in[17];

  char* p = (char*)d_ws;
  auto alloc = [&](size_t bytes) -> char* {
    char* r = p; p += (bytes + 255) & ~(size_t)255; return r;
  };
  __bf16* hbf0   = (__bf16*)alloc((size_t)BB * H * 2);
  __bf16* hbf1   = (__bf16*)alloc((size_t)BB * H * 2);
  float*  hf32   = (float*) alloc((size_t)BB * H * 4);
  __bf16* Wcat_t = (__bf16*)alloc((size_t)768 * KC * 2);
  __bf16* Wlat_t = (__bf16*)alloc((size_t)32 * 256 * 2);
  __bf16* W1t    = (__bf16*)alloc((size_t)512 * 32 * 2);
  __bf16* W2t    = (__bf16*)alloc((size_t)512 * 512 * 2);
  __bf16* W3t    = (__bf16*)alloc((size_t)32 * 512 * 2);
  __bf16* Wd1t   = (__bf16*)alloc((size_t)256 * 32 * 2);
  float*  z0f    = (float*) alloc((size_t)BB * L * 4);
  __bf16* z0b    = (__bf16*)alloc((size_t)BB * L * 2);
  float*  k1     = (float*) alloc((size_t)BB * L * 4);
  float*  k2     = (float*) alloc((size_t)BB * L * 4);
  float*  k3     = (float*) alloc((size_t)BB * L * 4);
  float*  k4     = (float*) alloc((size_t)BB * L * 4);
  float*  ztf    = (float*) alloc((size_t)BB * L * 4);
  __bf16* ztb    = (__bf16*)alloc((size_t)BB * L * 2);
  __bf16* t1b    = (__bf16*)alloc((size_t)BB * HO * 2);
  __bf16* t2b    = (__bf16*)alloc((size_t)BB * HO * 2);
  __bf16* d1b    = (__bf16*)alloc((size_t)BB * H * 2);

  prep_kernel<<<1024, 256, 0, stream>>>(W_ih, W_hh, W_lat, W1, W2, W3, Wd1,
                                        Wcat_t, Wlat_t, W1t, W2t, W3t, Wd1t,
                                        hbf0, hf32);

  // ---- GRU scan over reversed sequence: step s uses t_idx = T-1-s ----
  const size_t gru_lds = (size_t)(128 + 96) * GRU_STRIDE * 2;
  for (int s = 0; s < T; ++s) {
    const __bf16* hin  = (s & 1) ? hbf1 : hbf0;
    __bf16*       hout = (s & 1) ? hbf0 : hbf1;
    gru_step_kernel<<<dim3(BB / 128, H / 32), 128, gru_lds, stream>>>(
        x_hist, t_hist, Wcat_t, b_ih, b_hh, hin, hout, hf32, T - 1 - s);
  }
  const __bf16* hT = hbf0;  // T even -> final state lands in hbf0

  auto gemm = [&](const __bf16* A, const __bf16* Bt, const float* bias,
                  float* Cf, __bf16* Cb, int K, int N, int act) {
    size_t ldsb = (size_t)(128 + 32) * (K + 8) * 2;
    gemm_bf16_kernel<<<dim3(BB / 128, N / 32), 256, ldsb, stream>>>(A, Bt, bias, Cf, Cb, K, N, act);
  };

  // z0 = (hT @ W_lat + b_lat)[:, :L]
  gemm(hT, Wlat_t, b_lat, z0f, z0b, H, L, 0);

  auto feval = [&](const __bf16* zin, float* kout) {
    gemm(zin, W1t, b1, nullptr, t1b, L, HO, 1);   // tanh(z@W1+b1)
    gemm(t1b, W2t, b2, nullptr, t2b, HO, HO, 1);  // tanh(t1@W2+b2)
    gemm(t2b, W3t, b3, kout, nullptr, HO, L, 0);  // t2@W3+b3
  };

  const int nz = BB * L;
  auto combine = [&](const float* c1, const float* c2, const float* c3, const float* c4,
                     float a1, float a2, float a3, float a4) {
    combine_kernel<<<(nz + 255) / 256, 256, 0, stream>>>(z0f, c1, c2, c3, c4,
                                                         a1, a2, a3, a4, ztf, ztb, nz);
  };

  feval(z0b, k1);
  combine(k1, k1, k1, k1, DELTA / 3.0f, 0.0f, 0.0f, 0.0f);                    // za = z0 + h/3 k1
  feval(ztb, k2);
  combine(k1, k2, k1, k1, -DELTA / 3.0f, DELTA, 0.0f, 0.0f);                  // zb = z0 + h(k2 - k1/3)
  feval(ztb, k3);
  combine(k1, k2, k3, k1, DELTA, -DELTA, DELTA, 0.0f);                        // zc = z0 + h(k1 - k2 + k3)
  feval(ztb, k4);
  combine(k1, k2, k3, k4, DELTA / 8.0f, 3.0f * DELTA / 8.0f,
          3.0f * DELTA / 8.0f, DELTA / 8.0f);                                 // zT

  gemm(ztb, Wd1t, bd1, nullptr, d1b, L, H, 2);  // relu(zT@Wd1+bd1)
  decode2_kernel<<<(BB * OUTD + 255) / 256, 256, 0, stream>>>(d1b, Wd2, bd2, (float*)d_out);
}
